// MyGNN_82411832476044
// MI455X (gfx1250) — matrix operand order repaired and checked by reading, not compiled, not run on hardware
//
#include <hip/hip_runtime.h>
#include <hip/hip_bf16.h>
#include <math.h>

// ---------------------------------------------------------------------------
// GAT autoencoder for MI455X (gfx1250, wave32).
// Dense transforms / fc layers -> V_WMMA_F32_16X16X32_F16 with compile-time
// shapes so fragment loads vectorize to global_load_b128.
// Edge softmax/aggregation -> atomics (gather/scatter bound).
// ---------------------------------------------------------------------------

typedef __attribute__((ext_vector_type(16))) _Float16 v16h;
typedef __attribute__((ext_vector_type(8)))  float    v8f;

#define N_NODES  128000
#define N_EDGES  2000000
#define B_GRAPHS 64
#define NPG      2000
#define LATD     128

// order-preserving float<->uint mapping for atomicMax-based segment_max
__device__ __forceinline__ unsigned flt_ord(float f) {
    unsigned u = __float_as_uint(f);
    return (u & 0x80000000u) ? ~u : (u | 0x80000000u);
}
__device__ __forceinline__ float ord_flt(unsigned o) {
    return (o & 0x80000000u) ? __uint_as_float(o & 0x7FFFFFFFu)
                             : __uint_as_float(~o);
}

// ---------------------------------------------------------------------------
// WMMA GEMM:  Y[M, DOUT] = act( X[M, DIN] @ W[DOUT, DIN]^T + bias )
// One wave -> 16 rows x (16*NT) cols, K stepped by 32 (f16 in, f32 acc).
// Block = 128 threads = 4 waves -> 64 rows per block.
// grid.x = M/64, grid.y = ceil(DOUT / (16*NT)).
// Fragment packing per CDNA5 ISA 7.12.2 (wave32):
//   A: lanes 0-15 = rows, K 0..7 & 16..23 ; lanes 16-31 same rows, K 8..15 & 24..31
//   B: lane&15 = column; K as for A: lanes 0-15 K 0..7 & 16..23, lanes 16-31 K 8..15 & 24..31
//   C/D: lane&15 = column; VGPR v = row v (lanes 0-15) / row v+8 (lanes 16-31)
// ---------------------------------------------------------------------------
template<int DIN, int DOUT, int NT, int ACT, bool HASB>
__global__ __launch_bounds__(128)
void gemm_wmma(const float* __restrict__ X, const float* __restrict__ W,
               const float* __restrict__ bias, float* __restrict__ Y)
{
    constexpr int KPAD = (DIN + 31) & ~31;
    const int lane   = threadIdx.x & 31;
    const int wv     = threadIdx.x >> 5;
    const int half   = lane >> 4;     // K-half selector
    const int l16    = lane & 15;
    const int mtile  = blockIdx.x * 64 + wv * 16;
    const int ntile0 = blockIdx.y * (16 * NT);
    const int row    = mtile + l16;

    const float* __restrict__ xr = X + (size_t)row * DIN;

    v8f acc[NT];
    #pragma unroll
    for (int t = 0; t < NT; ++t) acc[t] = (v8f){};

    #pragma unroll
    for (int kk = 0; kk < KPAD; kk += 32) {
        // ---- A fragment (shared across all NT column tiles) ----
        v16h a;
        if constexpr (DIN == 3) {
            const float x0 = xr[0], x1 = xr[1], x2 = xr[2];
            #pragma unroll
            for (int i = 0; i < 8; ++i) {
                const int k0 = half * 8 + i;
                const float v = (k0 == 0) ? x0 : (k0 == 1) ? x1 : (k0 == 2) ? x2 : 0.0f;
                a[i]     = (_Float16)v;
                a[8 + i] = (_Float16)0.0f;
            }
        } else {
            #pragma unroll
            for (int i = 0; i < 8; ++i) {
                const int k0 = kk + half * 8 + i;        // folds: DIN % 8 == 0
                const int k1 = kk + 16 + half * 8 + i;
                a[i]     = (k0 < DIN) ? (_Float16)xr[k0] : (_Float16)0.0f;
                a[8 + i] = (k1 < DIN) ? (_Float16)xr[k1] : (_Float16)0.0f;
            }
        }

        // ---- NT B fragments + back-to-back WMMAs ----
        #pragma unroll
        for (int t = 0; t < NT; ++t) {
            const int  col = ntile0 + t * 16 + l16;
            const bool cok = (DOUT % 16 == 0) ? true : (col < DOUT);
            const float* __restrict__ wr =
                W + (size_t)((DOUT % 16 == 0) ? col : (cok ? col : 0)) * DIN;
            if (DOUT > 1024)  // streaming weight (fc_d, 131MB; L2-resident)
                __builtin_prefetch(wr + kk + 32, 0, 1);

            v16h b;
            if constexpr (DIN == 3) {
                const float w0 = wr[0], w1 = wr[1], w2 = wr[2];
                #pragma unroll
                for (int i = 0; i < 16; ++i) {
                    const int k = half * 16 + i;
                    const float v = (k == 0) ? w0 : (k == 1) ? w1 : (k == 2) ? w2 : 0.0f;
                    b[i] = (_Float16)(cok ? v : 0.0f);
                }
            } else {
                #pragma unroll
                for (int i = 0; i < 16; ++i) {
                    const int k = kk + ((i < 8) ? half * 8 + i : 16 + half * 8 + (i - 8));    // folds: DIN % 8 == 0
                    const float v = (k < DIN) ? wr[k] : 0.0f;
                    b[i] = (_Float16)(cok ? v : 0.0f);
                }
            }
            acc[t] = __builtin_amdgcn_wmma_f32_16x16x32_f16(
                /*neg_a=*/false, a, /*neg_b=*/false, b,
                /*c_mod=*/(short)0, acc[t], /*reuse_a=*/false, /*reuse_b=*/false);
        }
    }

    // ---- store D tiles ----
    #pragma unroll
    for (int t = 0; t < NT; ++t) {
        const int col = ntile0 + t * 16 + l16;
        if ((DOUT % 16 == 0) || col < DOUT) {
            #pragma unroll
            for (int v = 0; v < 8; ++v) {
                const int r = mtile + (half ? v + 8 : v);
                float y = acc[t][v];
                if constexpr (HASB) y += bias[col];
                if constexpr (ACT == 1) y = fmaxf(y, 0.0f);
                Y[(size_t)r * DOUT + col] = y;
            }
        }
    }
}

// ---------------------------------------------------------------------------
// Utility / edge-phase kernels
// ---------------------------------------------------------------------------
__global__ void zero_f(float* __restrict__ p, size_t n) {
    size_t i = (size_t)blockIdx.x * blockDim.x + threadIdx.x;
    if (i < n) p[i] = 0.0f;
}

// a_s[n] = h[n].a_src ; a_d[n] = h[n].a_dst
template<int DOUT>
__global__ void node_attn(const float* __restrict__ H,
                          const float* __restrict__ asrc,
                          const float* __restrict__ adst,
                          float* __restrict__ As, float* __restrict__ Ad)
{
    int n = blockIdx.x * blockDim.x + threadIdx.x;
    if (n >= N_NODES) return;
    const float* h = H + (size_t)n * DOUT;
    float s = 0.0f, d = 0.0f;
    #pragma unroll
    for (int c = 0; c < DOUT; ++c) { float v = h[c]; s += v * asrc[c]; d += v * adst[c]; }
    As[n] = s; Ad[n] = d;
}

// pass A: leaky_relu logit + segment_max(dst) via ordered-uint atomicMax
__global__ void edge_logit_max(const int* __restrict__ ei,
                               const float* __restrict__ As,
                               const float* __restrict__ Ad,
                               float* __restrict__ elog, unsigned* __restrict__ mo)
{
    int e = blockIdx.x * blockDim.x + threadIdx.x;
    if (e >= N_EDGES) return;
    int s = ei[e], d = ei[N_EDGES + e];
    float l = As[s] + Ad[d];
    l = (l > 0.0f) ? l : 0.2f * l;   // leaky_relu(0.2)
    elog[e] = l;
    atomicMax(&mo[d], flt_ord(l));
}

// decode ordered-uint max in place; non-finite (no incoming edges) -> 0
__global__ void decode_m(unsigned* __restrict__ mo) {
    int i = blockIdx.x * blockDim.x + threadIdx.x;
    if (i >= N_NODES) return;
    float f = ord_flt(mo[i]);
    if (!(fabsf(f) < INFINITY)) f = 0.0f;
    ((float*)mo)[i] = f;
}

// pass B: ez = exp(l - m[dst]); denom[dst] += ez ; elog <- ez
__global__ void edge_exp_sum(const int* __restrict__ ei,
                             float* __restrict__ elog,
                             const float* __restrict__ m,
                             float* __restrict__ denom)
{
    int e = blockIdx.x * blockDim.x + threadIdx.x;
    if (e >= N_EDGES) return;
    int d = ei[N_EDGES + e];
    float ez = __expf(elog[e] - m[d]);
    elog[e] = ez;
    atomicAdd(&denom[d], ez);
}

// pass C: Hout[dst] += alpha * Ht[src] ; 4 channels per thread
template<int DOUT>
__global__ void edge_aggregate(const int* __restrict__ ei,
                               const float* __restrict__ elog,
                               const float* __restrict__ denom,
                               const float* __restrict__ Ht,
                               float* __restrict__ Hout)
{
    constexpr int CHUNKS = (DOUT + 3) / 4;
    long long t = (long long)blockIdx.x * blockDim.x + threadIdx.x;
    long long total = (long long)N_EDGES * CHUNKS;
    if (t >= total) return;
    int e  = (int)(t / CHUNKS);
    int c0 = (int)(t % CHUNKS) * 4;
    int s = ei[e], d = ei[N_EDGES + e];
    float alpha = elog[e] / (denom[d] + 1e-16f);
    const float* hs = Ht + (size_t)s * DOUT;
    float* ho = Hout + (size_t)d * DOUT;
    #pragma unroll
    for (int i = 0; i < 4; ++i) {
        int c = c0 + i;
        if (c < DOUT) atomicAdd(&ho[c], alpha * hs[c]);
    }
}

// out = act(out + bias) ; act: 1 relu, 2 sigmoid
template<int DOUT, int ACT>
__global__ void finalize(float* __restrict__ H, const float* __restrict__ bias)
{
    long long t = (long long)blockIdx.x * blockDim.x + threadIdx.x;
    long long total = (long long)N_NODES * DOUT;
    if (t >= total) return;
    int c = (int)(t % DOUT);
    float v = H[t] + bias[c];
    if constexpr (ACT == 1)      v = fmaxf(v, 0.0f);
    else if constexpr (ACT == 2) v = 1.0f / (1.0f + __expf(-v));
    H[t] = v;
}

// pooled[b, c] = max over graph b's 2000 nodes of H[., 64]
__global__ void pool_max(const float* __restrict__ H, float* __restrict__ pooled)
{
    int t = blockIdx.x * blockDim.x + threadIdx.x;
    if (t >= B_GRAPHS * 64) return;
    int b = t >> 6, c = t & 63;
    const float* h = H + ((size_t)b * NPG) * 64 + c;
    float m = -INFINITY;
    for (int i = 0; i < NPG; ++i) m = fmaxf(m, h[(size_t)i * 64]);
    pooled[t] = m;
}

// ---------------------------------------------------------------------------
// Host-side orchestration
// ---------------------------------------------------------------------------
struct GatP { const float *W, *as, *ad, *b; };

template<int DIN, int DOUT, int NT, int ACT>
static void run_gat(const float* Hin, const GatP& p,
                    const int* ei, float* Ht, float* Hout,
                    float* As, float* Ad, unsigned* Mo, float* Den, float* Elog,
                    hipStream_t st)
{
    // h = Hin @ W^T  (WMMA)
    dim3 gg(N_NODES / 64, (DOUT + 16 * NT - 1) / (16 * NT));
    gemm_wmma<DIN, DOUT, NT, 0, false><<<gg, 128, 0, st>>>(Hin, p.W, nullptr, Ht);

    node_attn<DOUT><<<(N_NODES + 255) / 256, 256, 0, st>>>(Ht, p.as, p.ad, As, Ad);

    zero_f<<<(N_NODES + 255) / 256, 256, 0, st>>>((float*)Mo, N_NODES);
    zero_f<<<(N_NODES + 255) / 256, 256, 0, st>>>(Den, N_NODES);
    {
        size_t n = (size_t)N_NODES * DOUT;
        zero_f<<<(unsigned)((n + 255) / 256), 256, 0, st>>>(Hout, n);
    }

    edge_logit_max<<<(N_EDGES + 255) / 256, 256, 0, st>>>(ei, As, Ad, Elog, Mo);
    decode_m<<<(N_NODES + 255) / 256, 256, 0, st>>>(Mo);
    edge_exp_sum<<<(N_EDGES + 255) / 256, 256, 0, st>>>(ei, Elog, (const float*)Mo, Den);

    constexpr long long TOT = (long long)N_EDGES * ((DOUT + 3) / 4);
    edge_aggregate<DOUT><<<(unsigned)((TOT + 255) / 256), 256, 0, st>>>(
        ei, Elog, Den, Ht, Hout);

    constexpr long long FT = (long long)N_NODES * DOUT;
    finalize<DOUT, ACT><<<(unsigned)((FT + 255) / 256), 256, 0, st>>>(Hout, p.b);
}

extern "C" void kernel_launch(void* const* d_in, const int* in_sizes, int n_in,
                              void* d_out, int out_size, void* d_ws, size_t ws_size,
                              hipStream_t stream)
{
    (void)in_sizes; (void)n_in; (void)out_size; (void)ws_size;

    const float* x  = (const float*)d_in[0];
    const int*   ei = (const int*)d_in[1];
    // d_in[2] = batch (implied by node index; unused)

    int i = 3;
    GatP gp[6];  // g1e, g2e, g3e, g1d, g2d, g3d (dict insertion order)
    for (int L = 0; L < 6; ++L) {
        gp[L].W  = (const float*)d_in[i++];
        gp[L].as = (const float*)d_in[i++];
        gp[L].ad = (const float*)d_in[i++];
        gp[L].b  = (const float*)d_in[i++];
    }
    const float* fceW = (const float*)d_in[i++];
    const float* fceB = (const float*)d_in[i++];
    const float* fcdW = (const float*)d_in[i++];
    const float* fcdB = (const float*)d_in[i++];

    // workspace layout (floats)
    float* buf0 = (float*)d_ws;                    // N x 128
    float* buf1 = buf0 + (size_t)N_NODES * 128;    // N x 64
    float* htmp = buf1 + (size_t)N_NODES * 64;     // N x 64  (pre-aggregation h)
    float* As   = htmp + (size_t)N_NODES * 64;     // N
    float* Ad   = As + N_NODES;                    // N
    unsigned* Mo = (unsigned*)(Ad + N_NODES);      // N (ordered max -> float)
    float* Den  = (float*)(Mo + N_NODES);          // N
    float* Elog = Den + N_NODES;                   // E (logit -> ez)
    float* pooled = Elog + N_EDGES;                // 64 x 64
    float* z    = pooled + B_GRAPHS * 64;          // 64 x 128

    // ---- encoder ----
    run_gat<3,  16, 1, 1>(x,    gp[0], ei, htmp, buf1, As, Ad, Mo, Den, Elog, stream);
    run_gat<16, 32, 2, 1>(buf1, gp[1], ei, htmp, buf0, As, Ad, Mo, Den, Elog, stream);
    run_gat<32, 64, 4, 1>(buf0, gp[2], ei, htmp, buf1, As, Ad, Mo, Den, Elog, stream);

    pool_max<<<(B_GRAPHS * 64 + 255) / 256, 256, 0, stream>>>(buf1, pooled);

    // z = pooled @ fc_e_W^T + fc_e_b      [64,64] x [64->128]
    gemm_wmma<64, LATD, 8, 0, true><<<dim3(1, 1), 128, 0, stream>>>(
        pooled, fceW, fceB, z);

    // h0 = relu(z @ fc_d_W^T + fc_d_b)    [64,128] x [128->256000]
    // row-major [64,256000] is bit-identical to reshaped [128000,128]
    gemm_wmma<128, LATD * NPG, 1, 1, true><<<dim3(1, (LATD * NPG) / 16), 128, 0, stream>>>(
        z, fcdW, fcdB, buf0);

    // ---- decoder ----
    run_gat<128, 64, 4, 1>(buf0, gp[3], ei, htmp, buf1, As, Ad, Mo, Den, Elog, stream);
    run_gat<64,  16, 1, 1>(buf1, gp[4], ei, htmp, buf0, As, Ad, Mo, Den, Elog, stream);
    run_gat<16,  3,  1, 2>(buf0, gp[5], ei, htmp, (float*)d_out,
                           As, Ad, Mo, Den, Elog, stream);
}
